// EVAVisionTransformer_45724221833668
// MI455X (gfx1250) — compile-verified, matrix operand order checked by reading, not run on hardware
//
#include <hip/hip_runtime.h>
#include <hip/hip_bf16.h>

// ---- CDNA5 (gfx1250) wave32 WMMA implementation of 4-layer EVA ViT blocks ----

typedef __attribute__((ext_vector_type(16))) __bf16 v16bf;
typedef __attribute__((ext_vector_type(8)))  float  v8f;
typedef __attribute__((ext_vector_type(4)))  int    v4i;

union Frag16 { v16bf v; v4i i[2]; };

#define B_    16
#define N_    257
#define C_    1024
#define H_    16
#define D_    64
#define P_    77
#define M_    334      // P_ + N_
#define MP_   352      // padded to multiple of 32
#define L_    4
#define HID_  2730
#define HIDP_ 2816     // padded to multiple of 256
#define ROWS_  (B_*N_)   // 4112
#define TROWS_ (B_*P_)   // 1232

// --- CDNA5 async global->LDS path (ASYNCcnt), with safe fallback ---
#if defined(__has_builtin)
#  if __has_builtin(__builtin_amdgcn_global_load_async_to_lds_b128) && \
      __has_builtin(__builtin_amdgcn_s_wait_asynccnt)
#    define ASYNC_LDS 1
#  endif
#endif
#ifndef ASYNC_LDS
#  define ASYNC_LDS 0
#endif

#define AS1_ __attribute__((address_space(1)))
#define AS3_ __attribute__((address_space(3)))

__device__ __forceinline__ void cp16(__bf16* dst_lds, const __bf16* src_glb) {
#if ASYNC_LDS
  // prototype (from compiler diagnostic): (global int4*, lds int4*, imm, imm)
  __builtin_amdgcn_global_load_async_to_lds_b128(
      (AS1_ v4i*)src_glb, (AS3_ v4i*)dst_lds, 0, 0);
#else
  *(v4i*)dst_lds = *(const v4i*)src_glb;
#endif
}

// wait for all but `keep` outstanding async batches' ops (5 ops per stage call)
__device__ __forceinline__ void cp16_wait(int keep) {
#if ASYNC_LDS
  if (keep) __builtin_amdgcn_s_wait_asynccnt(5);
  else      __builtin_amdgcn_s_wait_asynccnt(0);
#else
  (void)keep;
#endif
}

__device__ __forceinline__ v8f wmma_bf16(const Frag16& a, const Frag16& b, v8f c) {
  return __builtin_amdgcn_wmma_f32_16x16x32_bf16(false, a.v, false, b.v,
                                                 (short)0, c, false, false);
}

// -------- rope tables --------
__global__ void rope_init_kernel(float* __restrict__ cosT, float* __restrict__ sinT) {
  int pos = blockIdx.x;         // 0..255
  int d   = threadIdx.x;        // 0..63
  int i   = d & 31;
  float inv = __powf(10000.f, -(float)(2 * i) / (float)D_);
  float ang = (float)pos * inv;
  cosT[pos * D_ + d] = __cosf(ang);
  sinT[pos * D_ + d] = __sinf(ang);
}

// -------- layernorm -> bf16 (padded width, pads zeroed) --------
__global__ __launch_bounds__(256) void ln_kernel(
    const float* __restrict__ in, int ldin,
    const float* __restrict__ w, const float* __restrict__ b,
    __bf16* __restrict__ out, int ldout, int width) {
  const int row = blockIdx.x;
  const float* x = in + (size_t)row * ldin;
  const int tid = threadIdx.x, lane = tid & 31, wid = tid >> 5;
  float s = 0.f, s2 = 0.f;
  for (int c = tid; c < width; c += 256) { float v = x[c]; s += v; s2 += v * v; }
  for (int o = 16; o > 0; o >>= 1) {
    s  += __shfl_down(s,  o, 32);
    s2 += __shfl_down(s2, o, 32);
  }
  __shared__ float rs[8], rs2[8];
  if (lane == 0) { rs[wid] = s; rs2[wid] = s2; }
  __syncthreads();
  if (tid == 0) {
    float a = 0.f, a2 = 0.f;
    for (int i = 0; i < 8; ++i) { a += rs[i]; a2 += rs2[i]; }
    rs[0] = a; rs2[0] = a2;
  }
  __syncthreads();
  float mean = rs[0] / (float)width;
  float var  = rs2[0] / (float)width - mean * mean;
  float rstd = rsqrtf(var + 1e-5f);
  for (int c = tid; c < ldout; c += 256)
    out[(size_t)row * ldout + c] =
        (c < width) ? (__bf16)((x[c] - mean) * rstd * w[c] + b[c]) : (__bf16)0.f;
}

// -------- weight fp32 [K,N] -> bf16 swizzled B-fragment tiles --------
__global__ void wswz_kernel(const float* __restrict__ W, int K, int N,
                            int Kp, int Np, __bf16* __restrict__ out) {
  long idx = (long)blockIdx.x * blockDim.x + threadIdx.x;
  long total = (long)Kp * Np;
  if (idx >= total) return;
  int n = (int)(idx % Np);
  int k = (int)(idx / Np);
  float val = (k < K && n < N) ? W[(size_t)k * N + n] : 0.f;
  int kk = k & 31, nn = n & 15;
  int lane, h;
  if      (kk < 8)  { lane = nn;      h = kk;            }
  else if (kk < 16) { lane = 16 + nn; h = kk - 8;        }
  else if (kk < 24) { lane = nn;      h = 8 + (kk - 16); }
  else              { lane = 16 + nn; h = 8 + (kk - 24); }
  int kt = k >> 5, nt = n >> 4;
  long dst = (((long)kt * (Np >> 4) + nt) * 32 + lane) * 16 + h;
  out[dst] = (__bf16)val;
}

// -------- stage one K-step of A (64x32) and B (16 tiles x 512) into LDS ------
__device__ __forceinline__ void stage_tiles(
    const __bf16* __restrict__ A, int ldA, int M,
    const __bf16* __restrict__ Bsw, int NT,
    int blockM, int ntBase, int kt, int tid,
    __bf16* As, __bf16* Bs) {
  int row = tid >> 2, part = tid & 3;
  int srow = blockM + row; if (srow > M - 1) srow = M - 1;
  cp16(&As[row * 32 + part * 8], &A[(size_t)srow * ldA + kt * 32 + part * 8]);
  const __bf16* bsrc = Bsw + ((size_t)kt * NT + ntBase) * 512;
#pragma unroll
  for (int i = 0; i < 4; ++i) {
    int ch = tid + i * 256;
    cp16(&Bs[ch * 8], &bsrc[ch * 8]);
  }
}

// -------- generic bf16 WMMA GEMM: out[map(r)] (+)= A@Bsw + bias --------
// block tile 64(M) x 256(N); 8 waves = 2(M) x 4(N); each wave 32x64 (8 WMMAs/kt)
// double-buffered LDS staging (async global->LDS when available).
__global__ __launch_bounds__(256) void gemm_wmma_kernel(
    const __bf16* __restrict__ A, int ldA,
    const __bf16* __restrict__ Bsw, int NT,
    float* __restrict__ out, int ldOut,
    const float* __restrict__ bias, int biasN,
    int M, int kTiles,
    int seg, int segStride, int segOff,
    int residual) {
  __shared__ __align__(16) __bf16 As[2][64 * 32];
  __shared__ __align__(16) __bf16 Bs[2][16 * 512];
  const int tid = threadIdx.x;
  const int wid = tid >> 5, lane = tid & 31;
  const int l16 = lane & 15, hi = lane >> 4;
  const int waveM = wid & 1, waveN = wid >> 1;
  const int blockM = blockIdx.y * 64;
  const int ntBase = blockIdx.x * 16;

  v8f acc[2][4] = {};

  stage_tiles(A, ldA, M, Bsw, NT, blockM, ntBase, 0, tid, As[0], Bs[0]);
  int buf = 0;
  for (int kt = 0; kt < kTiles; ++kt) {
    int haveNext = (kt + 1 < kTiles);
    if (haveNext)
      stage_tiles(A, ldA, M, Bsw, NT, blockM, ntBase, kt + 1, tid,
                  As[buf ^ 1], Bs[buf ^ 1]);
    cp16_wait(haveNext);   // drain batch kt (keep kt+1 in flight)
    __syncthreads();

    Frag16 a[2];
#pragma unroll
    for (int am = 0; am < 2; ++am) {
      const __bf16* ab = &As[buf][(waveM * 32 + am * 16 + l16) * 32 + hi * 8];
      a[am].i[0] = *(const v4i*)ab;
      a[am].i[1] = *(const v4i*)(ab + 16);
    }
#pragma unroll
    for (int t = 0; t < 4; ++t) {
      Frag16 bf;
      const __bf16* bb = &Bs[buf][(waveN * 4 + t) * 512 + lane * 16];
      bf.i[0] = *(const v4i*)bb;
      bf.i[1] = *(const v4i*)(bb + 8);
      acc[0][t] = wmma_bf16(a[0], bf, acc[0][t]);
      acc[1][t] = wmma_bf16(a[1], bf, acc[1][t]);
    }
    __syncthreads();   // protect buf^1 from being overwritten next iteration
    buf ^= 1;
  }

  const int colBase = blockIdx.x * 256 + waveN * 64;
#pragma unroll
  for (int am = 0; am < 2; ++am) {
    int rowBase = blockM + waveM * 32 + am * 16 + hi * 8;
#pragma unroll
    for (int t = 0; t < 4; ++t) {
      int col = colBase + t * 16 + l16;
      float bvv = 0.f;
      if (bias != nullptr && col < biasN) bvv = bias[col];
#pragma unroll
      for (int r = 0; r < 8; ++r) {
        int row = rowBase + r;
        if (row < M) {
          size_t dst = (size_t)((row / seg) * segStride + segOff + row % seg) * ldOut + col;
          float v = acc[am][t][r] + bvv;
          if (residual) v += out[dst];
          out[dst] = v;
        }
      }
    }
  }
}

// -------- q pack: rope (skip CLS) + bf16 [B,H,N,D] --------
__global__ void pack_q_kernel(const float* __restrict__ q,
                              const float* __restrict__ cosT,
                              const float* __restrict__ sinT,
                              __bf16* __restrict__ qbf) {
  int idx = blockIdx.x * blockDim.x + threadIdx.x;
  if (idx >= B_ * H_ * N_) return;
  int n = idx % N_; int bh = idx / N_;
  int h = bh % H_;  int b = bh / H_;
  const float* src = q + ((size_t)(b * N_ + n)) * C_ + h * D_;
  __bf16* dst = qbf + (size_t)idx * D_;
  if (n == 0) {
    for (int d = 0; d < D_; ++d) dst[d] = (__bf16)src[d];
  } else {
    int pos = n - 1;
    for (int d = 0; d < D_; ++d) {
      float t = src[d];
      float rot = (d < 32) ? -src[d + 32] : src[d - 32];
      dst[d] = (__bf16)(t * cosT[pos * D_ + d] + rot * sinT[pos * D_ + d]);
    }
  }
}

// -------- kv pack: rope on k (skip prefix+CLS), k->[B,H,MP,D], v->[B,H,D,MP] --------
__global__ void pack_kv_kernel(const float* __restrict__ k, const float* __restrict__ v,
                               const float* __restrict__ cosT, const float* __restrict__ sinT,
                               __bf16* __restrict__ kbf, __bf16* __restrict__ vtbf) {
  int idx = blockIdx.x * blockDim.x + threadIdx.x;
  if (idx >= B_ * H_ * MP_) return;
  int m = idx % MP_; int bh = idx / MP_;
  int h = bh % H_;   int b = bh / H_;
  __bf16* kd = kbf + (size_t)idx * D_;
  __bf16* vt = vtbf + (size_t)bh * D_ * MP_;
  if (m >= M_) {
    for (int d = 0; d < D_; ++d) { kd[d] = (__bf16)0.f; vt[(size_t)d * MP_ + m] = (__bf16)0.f; }
    return;
  }
  const float* ks = k + ((size_t)(b * M_ + m)) * C_ + h * D_;
  const float* vs = v + ((size_t)(b * M_ + m)) * C_ + h * D_;
  for (int d = 0; d < D_; ++d) vt[(size_t)d * MP_ + m] = (__bf16)vs[d];
  if (m < P_ + 1) {
    for (int d = 0; d < D_; ++d) kd[d] = (__bf16)ks[d];
  } else {
    int pos = m - (P_ + 1);
    for (int d = 0; d < D_; ++d) {
      float t = ks[d];
      float rot = (d < 32) ? -ks[d + 32] : ks[d - 32];
      kd[d] = (__bf16)(t * cosT[pos * D_ + d] + rot * sinT[pos * D_ + d]);
    }
  }
}

// -------- attention: one wave per (b,h, 16-query tile) --------
__global__ __launch_bounds__(32) void attn_kernel(
    const __bf16* __restrict__ qbf, const __bf16* __restrict__ kbf,
    const __bf16* __restrict__ vtbf, const float* __restrict__ gate,
    float* __restrict__ o) {
  __shared__ __align__(16) float  sS[16 * MP_];
  __shared__ __align__(16) __bf16 sP[16 * MP_];
  const int nt = blockIdx.x % 17;
  const int bh = blockIdx.x / 17;
  const int h = bh % H_, bIdx = bh / H_;
  const int lane = threadIdx.x & 31, l16 = lane & 15, hi = lane >> 4;
  const int n0 = nt * 16;
  const float g = tanhf(gate[h]);

  // q fragments (two K=32 chunks over D=64)
  Frag16 qa[2];
  {
    int n = n0 + l16; if (n > N_ - 1) n = N_ - 1;
    const __bf16* qb = qbf + ((size_t)bh * N_ + n) * D_;
#pragma unroll
    for (int kc = 0; kc < 2; ++kc) {
      const __bf16* p = qb + kc * 32 + hi * 8;
      qa[kc].i[0] = *(const v4i*)p;
      qa[kc].i[1] = *(const v4i*)(p + 16);
    }
  }

  // scores -> LDS
  for (int mt = 0; mt < MP_ / 16; ++mt) {
    v8f sa = {};
#pragma unroll
    for (int kc = 0; kc < 2; ++kc) {
      Frag16 kb;
      const __bf16* p = kbf + ((size_t)bh * MP_ + mt * 16 + l16) * D_ + kc * 32 + hi * 8;
      kb.i[0] = *(const v4i*)p;
      kb.i[1] = *(const v4i*)(p + 16);
      sa = wmma_bf16(qa[kc], kb, sa);
    }
#pragma unroll
    for (int r = 0; r < 8; ++r)
      sS[(r + 8 * hi) * MP_ + mt * 16 + l16] = sa[r] * 0.125f;  // 1/sqrt(64)
  }
  __syncthreads();

  // softmax over columns 1..M_-1; col0 = tanh(gate) (softmax over 1 elem == 1)
  if (lane < 16) {
    float* srow = &sS[lane * MP_];
    float mx = -1e30f;
    for (int m = 1; m < M_; ++m) mx = fmaxf(mx, srow[m]);
    float sum = 0.f;
    for (int m = 1; m < M_; ++m) { float e = __expf(srow[m] - mx); srow[m] = e; sum += e; }
    float inv = 1.f / sum;
    __bf16* prow = &sP[lane * MP_];
    prow[0] = (__bf16)g;
    for (int m = 1; m < M_; ++m)  prow[m] = (__bf16)(srow[m] * inv);
    for (int m = M_; m < MP_; ++m) prow[m] = (__bf16)0.f;
  }
  __syncthreads();

  // o = probs @ v
  v8f oacc[4] = {};
  for (int mc = 0; mc < MP_ / 32; ++mc) {
    Frag16 pa;
    const __bf16* pp = &sP[l16 * MP_ + mc * 32 + hi * 8];
    pa.i[0] = *(const v4i*)pp;
    pa.i[1] = *(const v4i*)(pp + 16);
#pragma unroll
    for (int ds = 0; ds < 4; ++ds) {
      Frag16 vb;
      const __bf16* p = vtbf + ((size_t)bh * D_ + ds * 16 + l16) * MP_ + mc * 32 + hi * 8;
      vb.i[0] = *(const v4i*)p;
      vb.i[1] = *(const v4i*)(p + 16);
      oacc[ds] = wmma_bf16(pa, vb, oacc[ds]);
    }
  }
#pragma unroll
  for (int ds = 0; ds < 4; ++ds)
#pragma unroll
    for (int r = 0; r < 8; ++r) {
      int n = n0 + r + 8 * hi;
      if (n < N_)
        o[((size_t)(bIdx * N_ + n)) * C_ + h * D_ + ds * 16 + l16] = oacc[ds][r];
    }
}

// -------- SwiGLU elementwise: a = silu(a) * b --------
__global__ void swiglu_kernel(float* __restrict__ a, const float* __restrict__ b, long n) {
  long i = (long)blockIdx.x * blockDim.x + threadIdx.x;
  if (i >= n) return;
  float x = a[i];
  a[i] = (x / (1.f + __expf(-x))) * b[i];
}

extern "C" void kernel_launch(void* const* d_in, const int* in_sizes, int n_in,
                              void* d_out, int out_size, void* d_ws, size_t ws_size,
                              hipStream_t stream) {
  (void)in_sizes; (void)n_in; (void)out_size; (void)ws_size;
  const float* x      = (const float*)d_in[0];
  const float* x_text = (const float*)d_in[1];
  const float* n1w = (const float*)d_in[2];
  const float* n1b = (const float*)d_in[3];
  const float* n2w = (const float*)d_in[4];
  const float* n2b = (const float*)d_in[5];
  const float* n3w = (const float*)d_in[6];
  const float* n3b = (const float*)d_in[7];
  const float* Wq  = (const float*)d_in[8];
  const float* bq  = (const float*)d_in[9];
  const float* Wk  = (const float*)d_in[10];
  const float* Wv  = (const float*)d_in[11];
  const float* bv  = (const float*)d_in[12];
  const float* gate= (const float*)d_in[13];
  const float* liw = (const float*)d_in[14];
  const float* lib = (const float*)d_in[15];
  const float* Wo  = (const float*)d_in[16];
  const float* bo  = (const float*)d_in[17];
  const float* W1  = (const float*)d_in[18];
  const float* b1  = (const float*)d_in[19];
  const float* W2  = (const float*)d_in[20];
  const float* b2  = (const float*)d_in[21];
  const float* fw  = (const float*)d_in[22];
  const float* fb  = (const float*)d_in[23];
  const float* W3  = (const float*)d_in[24];
  const float* b3  = (const float*)d_in[25];

  char* ws = (char*)d_ws;
  size_t off = 0;
  auto alloc = [&](size_t bytes) -> char* {
    char* p = ws + off;
    off += (bytes + 255) & ~(size_t)255;
    return p;
  };

  float*  ropeC = (float*)alloc((size_t)256 * D_ * 4);
  float*  ropeS = (float*)alloc((size_t)256 * D_ * 4);
  __bf16* hbf   = (__bf16*)alloc((size_t)ROWS_ * C_ * 2);
  __bf16* xtbf  = (__bf16*)alloc((size_t)TROWS_ * C_ * 2);
  float*  qf    = (float*)alloc((size_t)ROWS_ * C_ * 4);
  float*  kf    = (float*)alloc((size_t)B_ * M_ * C_ * 4);
  float*  vf    = (float*)alloc((size_t)B_ * M_ * C_ * 4);
  __bf16* qbf   = (__bf16*)alloc((size_t)B_ * H_ * N_ * D_ * 2);
  __bf16* kbf   = (__bf16*)alloc((size_t)B_ * H_ * MP_ * D_ * 2);
  __bf16* vtbf  = (__bf16*)alloc((size_t)B_ * H_ * D_ * MP_ * 2);
  float*  obuf  = (float*)alloc((size_t)ROWS_ * C_ * 4);
  __bf16* olnbf = (__bf16*)alloc((size_t)ROWS_ * C_ * 2);
  float*  hid1  = (float*)alloc((size_t)ROWS_ * HIDP_ * 4);
  float*  hid2  = (float*)alloc((size_t)ROWS_ * HIDP_ * 4);
  __bf16* hlnbf = (__bf16*)alloc((size_t)ROWS_ * HIDP_ * 2);
  __bf16* wq_s  = (__bf16*)alloc((size_t)C_ * C_ * 2);
  __bf16* wk_s  = (__bf16*)alloc((size_t)C_ * C_ * 2);
  __bf16* wv_s  = (__bf16*)alloc((size_t)C_ * C_ * 2);
  __bf16* wo_s  = (__bf16*)alloc((size_t)C_ * C_ * 2);
  __bf16* w1_s  = (__bf16*)alloc((size_t)C_ * HIDP_ * 2);
  __bf16* w2_s  = (__bf16*)alloc((size_t)C_ * HIDP_ * 2);
  __bf16* w3_s  = (__bf16*)alloc((size_t)HIDP_ * C_ * 2);

  float* xc = (float*)d_out;
  (void)hipMemcpyAsync(xc, x, (size_t)ROWS_ * C_ * 4, hipMemcpyDeviceToDevice, stream);
  rope_init_kernel<<<256, 64, 0, stream>>>(ropeC, ropeS);

  const dim3 gC(C_ / 256, (ROWS_ + 63) / 64);      // 4 x 65
  const dim3 gCt(C_ / 256, (TROWS_ + 63) / 64);    // 4 x 20
  const dim3 gHID(HIDP_ / 256, (ROWS_ + 63) / 64); // 11 x 65

  for (int l = 0; l < L_; ++l) {
    const float* Wq_l = Wq + (size_t)l * C_ * C_;
    const float* Wk_l = Wk + (size_t)l * C_ * C_;
    const float* Wv_l = Wv + (size_t)l * C_ * C_;
    const float* Wo_l = Wo + (size_t)l * C_ * C_;
    const float* W1_l = W1 + (size_t)l * C_ * HID_;
    const float* W2_l = W2 + (size_t)l * C_ * HID_;
    const float* W3_l = W3 + (size_t)l * HID_ * C_;

    long tcc = (long)C_ * C_;
    wswz_kernel<<<(tcc + 255) / 256, 256, 0, stream>>>(Wq_l, C_, C_, C_, C_, wq_s);
    wswz_kernel<<<(tcc + 255) / 256, 256, 0, stream>>>(Wk_l, C_, C_, C_, C_, wk_s);
    wswz_kernel<<<(tcc + 255) / 256, 256, 0, stream>>>(Wv_l, C_, C_, C_, C_, wv_s);
    wswz_kernel<<<(tcc + 255) / 256, 256, 0, stream>>>(Wo_l, C_, C_, C_, C_, wo_s);
    long tch = (long)C_ * HIDP_;
    wswz_kernel<<<(tch + 255) / 256, 256, 0, stream>>>(W1_l, C_, HID_, C_, HIDP_, w1_s);
    wswz_kernel<<<(tch + 255) / 256, 256, 0, stream>>>(W2_l, C_, HID_, C_, HIDP_, w2_s);
    long thc = (long)HIDP_ * C_;
    wswz_kernel<<<(thc + 255) / 256, 256, 0, stream>>>(W3_l, HID_, C_, HIDP_, C_, w3_s);

    // h = ln1(xc); xt = ln3(x_text)
    ln_kernel<<<ROWS_, 256, 0, stream>>>(xc, C_, n1w + l * C_, n1b + l * C_, hbf, C_, C_);
    ln_kernel<<<TROWS_, 256, 0, stream>>>(x_text, C_, n3w + l * C_, n3b + l * C_, xtbf, C_, C_);

    // q = h@Wq + bq
    gemm_wmma_kernel<<<gC, 256, 0, stream>>>(hbf, C_, wq_s, C_ / 16, qf, C_,
                                             bq + l * C_, C_, ROWS_, C_ / 32,
                                             ROWS_, 0, 0, 0);
    // k = [xt@Wk ; h@Wk] interleaved per batch into [B, 334, C]
    gemm_wmma_kernel<<<gCt, 256, 0, stream>>>(xtbf, C_, wk_s, C_ / 16, kf, C_,
                                              nullptr, 0, TROWS_, C_ / 32,
                                              P_, M_, 0, 0);
    gemm_wmma_kernel<<<gC, 256, 0, stream>>>(hbf, C_, wk_s, C_ / 16, kf, C_,
                                             nullptr, 0, ROWS_, C_ / 32,
                                             N_, M_, P_, 0);
    // v = [xt@Wv+bv ; h@Wv+bv]
    gemm_wmma_kernel<<<gCt, 256, 0, stream>>>(xtbf, C_, wv_s, C_ / 16, vf, C_,
                                              bv + l * C_, C_, TROWS_, C_ / 32,
                                              P_, M_, 0, 0);
    gemm_wmma_kernel<<<gC, 256, 0, stream>>>(hbf, C_, wv_s, C_ / 16, vf, C_,
                                             bv + l * C_, C_, ROWS_, C_ / 32,
                                             N_, M_, P_, 0);

    pack_q_kernel<<<(B_ * H_ * N_ + 255) / 256, 256, 0, stream>>>(qf, ropeC, ropeS, qbf);
    pack_kv_kernel<<<(B_ * H_ * MP_ + 255) / 256, 256, 0, stream>>>(kf, vf, ropeC, ropeS,
                                                                    kbf, vtbf);

    attn_kernel<<<B_ * H_ * 17, 32, 0, stream>>>(qbf, kbf, vtbf, gate + l * H_, obuf);

    // inner_attn_ln, residual 1
    ln_kernel<<<ROWS_, 256, 0, stream>>>(obuf, C_, liw + l * C_, lib + l * C_, olnbf, C_, C_);
    gemm_wmma_kernel<<<gC, 256, 0, stream>>>(olnbf, C_, wo_s, C_ / 16, xc, C_,
                                             bo + l * C_, C_, ROWS_, C_ / 32,
                                             ROWS_, 0, 0, 1);

    // FFN
    ln_kernel<<<ROWS_, 256, 0, stream>>>(xc, C_, n2w + l * C_, n2b + l * C_, hbf, C_, C_);
    gemm_wmma_kernel<<<gHID, 256, 0, stream>>>(hbf, C_, w1_s, HIDP_ / 16, hid1, HIDP_,
                                               b1 + (size_t)l * HID_, HID_, ROWS_, C_ / 32,
                                               ROWS_, 0, 0, 0);
    gemm_wmma_kernel<<<gHID, 256, 0, stream>>>(hbf, C_, w2_s, HIDP_ / 16, hid2, HIDP_,
                                               b2 + (size_t)l * HID_, HID_, ROWS_, C_ / 32,
                                               ROWS_, 0, 0, 0);
    long nsw = (long)ROWS_ * HIDP_;
    swiglu_kernel<<<(nsw + 255) / 256, 256, 0, stream>>>(hid1, hid2, nsw);
    ln_kernel<<<ROWS_, 256, 0, stream>>>(hid1, HIDP_, fw + (size_t)l * HID_,
                                         fb + (size_t)l * HID_, hlnbf, HIDP_, HID_);
    gemm_wmma_kernel<<<gC, 256, 0, stream>>>(hlnbf, HIDP_, w3_s, C_ / 16, xc, C_,
                                             b3 + l * C_, C_, ROWS_, HIDP_ / 32,
                                             ROWS_, 0, 0, 1);
  }
}
